// BiLSTM_CRF_70677981823217
// MI455X (gfx1250) — matrix operand order repaired and checked by reading, not compiled, not run on hardware
//
#include <hip/hip_runtime.h>
#include <hip/hip_bf16.h>

typedef __attribute__((ext_vector_type(16))) __bf16        v16bf;
typedef __attribute__((ext_vector_type(8)))  unsigned int  v8u;
typedef __attribute__((ext_vector_type(8)))  float         v8f;
typedef __attribute__((ext_vector_type(4)))  unsigned int  u32x4;
typedef __attribute__((ext_vector_type(8)))  int           i32x8;
typedef __attribute__((ext_vector_type(4)))  int           i32x4;

#define SEQ   4096
#define EMB   512
#define H     512
#define G4    2048   // 4*H
#define HID   1024
#define TAGS  24
#define START_TAG 22
#define STOP_TAG  23
#define NEGV  (-10000.0f)

// round-to-nearest-even f32->bf16 for two values, packed with one v_perm_b32
__device__ __forceinline__ unsigned pack2bf(float f0, float f1) {
    unsigned u0 = __float_as_uint(f0), u1 = __float_as_uint(f1);
    u0 = u0 + 0x7FFFu + ((u0 >> 16) & 1u);
    u1 = u1 + 0x7FFFu + ((u1 >> 16) & 1u);
    // result bytes [r1.b3 r1.b2 r0.b3 r0.b2]
    return __builtin_amdgcn_perm(u1, u0, 0x07060302u);
}

__device__ __forceinline__ float sigf(float x) {
    return 1.0f / (1.0f + __expf(-x));
}

// pack a 16-element bf16 WMMA operand from four float4 runs (pair r -> u[r])
__device__ __forceinline__ v16bf pack_tile(float4 a, float4 b, float4 c, float4 d) {
    v8u u;
    u[0] = pack2bf(a.x, a.y); u[1] = pack2bf(a.z, a.w);
    u[2] = pack2bf(b.x, b.y); u[3] = pack2bf(b.z, b.w);
    u[4] = pack2bf(c.x, c.y); u[5] = pack2bf(c.z, c.w);
    u[6] = pack2bf(d.x, d.y); u[7] = pack2bf(d.z, d.w);
    return __builtin_bit_cast(v16bf, u);
}

// ---------------------------------------------------------------------------
// Kernel A: embedding gather + input projection, both directions.
// One wave computes a 16x64 strip of xg = emb[sentence] @ w_ih^T + (b_ih+b_hh):
// the A tile (16x32 bf16) is loaded/converted once per K-chunk and reused by
// 4 WMMAs (4 N-tiles). All global traffic is b128.
// Waves: dir(2) x mt(256) x ntb(32) = 16384 -> 2048 blocks x 256.
// ---------------------------------------------------------------------------
__global__ void __launch_bounds__(256) embed_proj_kernel(
    const int* __restrict__ sent, const float* __restrict__ emb,
    const float* __restrict__ w_ih_f, const float* __restrict__ b_ih_f, const float* __restrict__ b_hh_f,
    const float* __restrict__ w_ih_b, const float* __restrict__ b_ih_b, const float* __restrict__ b_hh_b,
    float* __restrict__ xg_f, float* __restrict__ xg_b)
{
    int wave = blockIdx.x * 8 + (threadIdx.x >> 5);
    int lane = threadIdx.x & 31;
    int dir  = wave >> 13;
    int tid  = wave & 8191;
    int mt = tid >> 5, ntb = tid & 31;
    const float* w_ih = dir ? w_ih_b : w_ih_f;
    const float* b_ih = dir ? b_ih_b : b_ih_f;
    const float* b_hh = dir ? b_hh_b : b_hh_f;
    float*       xg   = dir ? xg_b   : xg_f;

    int m0 = mt * 16, n0 = ntb * 64;
    int Ml    = lane & 15;
    int khalf = (lane >> 4) & 1;

    const float* arow = emb + (long)sent[m0 + Ml] * EMB;   // gathered x row
    const float* brow[4];
    #pragma unroll
    for (int s = 0; s < 4; ++s)
        brow[s] = w_ih + (long)(n0 + s * 16 + Ml) * EMB;   // B[k][n] = w_ih[n][k]

    v8f acc[4] = {};
    for (int kt = 0; kt < 16; ++kt) {
        int kbase = kt * 32;
        // A 16x32 bf16 layout: one contiguous 16-float run per lane
        const float* a0 = arow + kbase + khalf * 16;       // K group 0
        const float* a1 = a0 + 8;                          // K group 1
        v16bf A = pack_tile(*(const float4*)a0, *(const float4*)(a0 + 4),
                            *(const float4*)a1, *(const float4*)(a1 + 4));
        #pragma unroll
        for (int s = 0; s < 4; ++s) {
            // B 32x16 bf16 layout: one contiguous 16-float run per lane
            const float* bp = brow[s] + kbase + khalf * 16;
            v16bf B = pack_tile(*(const float4*)bp,        *(const float4*)(bp + 4),
                                *(const float4*)(bp + 8),  *(const float4*)(bp + 12));
            acc[s] = __builtin_amdgcn_wmma_f32_16x16x32_bf16(
                         false, A, false, B, (short)0, acc[s], false, false);
        }
    }
    #pragma unroll
    for (int s = 0; s < 4; ++s) {
        int n = n0 + s * 16 + Ml;
        float bias = b_ih[n] + b_hh[n];
        #pragma unroll
        for (int v = 0; v < 8; ++v) {
            int m = m0 + v + khalf * 8;      // C/D layout: VGPR v -> M=v (+8 hi lanes)
            xg[(long)m * G4 + n] = acc[s][v] + bias;
        }
    }
}

// ---------------------------------------------------------------------------
// Kernel B: serial LSTM scan, both directions.
// 32 WGs x 256 threads; WGs [0..15] forward, [16..31] backward. Each WG owns
// 32 hidden units (rows {u,u+512,u+1024,u+1536} of W_hh) and keeps its
// 128x512 f32 slice resident in ~266KB of CDNA5 LDS (impossible on 64KB CDNA4).
// The slice is staged by the Tensor Data Mover (4 2D-tile descriptors, LDS
// padding 4 dwords / 256 dwords -> 520-float row stride, bank-conflict free).
// Per step: local matvec from LDS, h published through L2 (agent scope),
// 16-WG counter barrier, h re-broadcast.
// ---------------------------------------------------------------------------
__global__ void __launch_bounds__(256) lstm_scan_kernel(
    const float* __restrict__ xg_f, const float* __restrict__ xg_b,
    const float* __restrict__ w_hh_f, const float* __restrict__ w_hh_b,
    const float* __restrict__ h0, const float* __restrict__ c0,
    float* __restrict__ hf, float* __restrict__ hb,
    float* __restrict__ h_glob, int* __restrict__ bar)
{
    extern __shared__ float smem[];
    const int WSTR = 520;                  // 256 | 4 pad | 256 | 4 pad
    float* W_s = smem;                     // 128 * 520 (at dynamic-LDS base 0)
    float* h_s = smem + 128 * WSTR;        // 512
    float* g_s = h_s + H;                  // 128

    int dir = blockIdx.x >> 4;
    int wgi = blockIdx.x & 15;
    int U0  = wgi * 32;
    int j   = threadIdx.x;

    const float* xg = dir ? xg_b   : xg_f;
    const float* W  = dir ? w_hh_b : w_hh_f;
    float*       hs = dir ? hb     : hf;

#if __has_builtin(__builtin_amdgcn_tensor_load_to_lds)
    if (j < 32) {   // wave 0 issues 4 TDM tile loads (one per gate block)
        #pragma unroll
        for (int gate = 0; gate < 4; ++gate) {
            unsigned lds_base = (unsigned)(gate * 32 * WSTR * 4);
            unsigned long long ga =
                (unsigned long long)(uintptr_t)(W + ((long)gate * 512 + U0) * H);
            u32x4 g0; i32x8 g1;
            i32x4 gz4 = {0, 0, 0, 0};
            i32x8 gz8 = {0, 0, 0, 0, 0, 0, 0, 0};
            g0[0] = 1u;                                   // count=1, user descriptor
            g0[1] = lds_base;                             // lds_addr (bytes)
            g0[2] = (unsigned)ga;                         // global_addr[31:0]
            g0[3] = (unsigned)((ga >> 32) & 0x01FFFFFFull) | (2u << 30); // [56:32]|type=2
            // data_size=4B(2), pad_enable, pad_interval=256dw(7), pad_amount=4dw(3)
            g1[0] = (2 << 16) | (1 << 20) | (7 << 22) | (3 << 25);
            g1[1] = (int)(512u << 16);                    // tensor_dim0[15:0]
            g1[2] = (int)(32u << 16);                     // dim0 hi=0 | tensor_dim1=32
            g1[3] = (int)(512u << 16);                    // dim1 hi=0 | tile_dim0=512
            g1[4] = 32;                                   // tile_dim1=32, tile_dim2=0
            g1[5] = 512;                                  // tensor_dim0_stride
            g1[6] = 0;
            g1[7] = 0;
            __builtin_amdgcn_tensor_load_to_lds(g0, g1, gz4, gz4, gz8, 0);
        }
        __builtin_amdgcn_s_wait_tensorcnt(0);
    }
#else
    for (int idx = j; idx < 128 * 128; idx += 256) {      // float4 granularity
        int lr = idx >> 7, c4 = idx & 127;
        int gr = ((lr >> 5) << 9) + U0 + (lr & 31);
        float4 v = ((const float4*)(W + (long)gr * H))[c4];
        int off = c4 * 4 + (c4 >= 64 ? 4 : 0);            // mid-row 4-dword gap
        *(float4*)(W_s + lr * WSTR + off) = v;
    }
#endif
    for (int idx = j; idx < H; idx += 256) h_s[idx] = h0[dir * H + idx];
    float c = (j < 32) ? c0[dir * H + U0 + j] : 0.0f;
    __syncthreads();

    int lr   = j >> 1;                 // local row 0..127
    int half = j & 1;                  // 512-dot split across lane pairs
    const float4* Wp = (const float4*)(W_s + lr * WSTR + (half ? 260 : 0));

    for (int s = 0; s < SEQ; ++s) {
        int t = dir ? (SEQ - 1 - s) : s;
        const float* xgt = xg + (long)t * G4;
        __builtin_prefetch(xg + (long)(dir ? (t > 0 ? t - 1 : 0)
                                           : (t < SEQ - 1 ? t + 1 : t)) * G4
                              + ((lr >> 5) << 9) + U0, 0, 0);

        const float4* Hp = (const float4*)(h_s + half * 256);
        float d = 0.0f;
        #pragma unroll 8
        for (int k4 = 0; k4 < 64; ++k4) {
            float4 w = Wp[k4], h = Hp[k4];
            d += w.x * h.x + w.y * h.y + w.z * h.z + w.w * h.w;
        }
        d += __shfl_xor(d, 1);
        if (half == 0) {
            int gr = ((lr >> 5) << 9) + U0 + (lr & 31);
            g_s[lr] = xgt[gr] + d;
        }
        __syncthreads();

        if (j < 32) {                  // one thread per owned unit
            float ig = sigf(g_s[j]);
            float fg = sigf(g_s[32 + j]);
            float gg = tanhf(g_s[64 + j]);
            float og = sigf(g_s[96 + j]);
            c = fg * c + ig * gg;
            float h = og * tanhf(c);
            hs[(long)t * H + U0 + j] = h;
            __hip_atomic_store(&h_glob[dir * H + U0 + j], h,
                               __ATOMIC_RELEASE, __HIP_MEMORY_SCOPE_AGENT);
        }
        __syncthreads();

        if (j == 0) {                  // 16-WG barrier, monotonic counter
            __hip_atomic_fetch_add(&bar[dir], 1, __ATOMIC_ACQ_REL,
                                   __HIP_MEMORY_SCOPE_AGENT);
            while (__hip_atomic_load(&bar[dir], __ATOMIC_ACQUIRE,
                                     __HIP_MEMORY_SCOPE_AGENT) < 16 * (s + 1)) {
                __builtin_amdgcn_s_sleep(1);
            }
        }
        __syncthreads();

        for (int idx = j; idx < H; idx += 256) {
            h_s[idx] = __hip_atomic_load(&h_glob[dir * H + idx],
                                         __ATOMIC_ACQUIRE, __HIP_MEMORY_SCOPE_AGENT);
        }
        __syncthreads();
    }
}

// ---------------------------------------------------------------------------
// Kernel C: tag projection feats = [hf|hb] @ w_tag^T + b_tag.
// One wave computes a 16x32 strip (N padded 24->32, invalid columns zeroed),
// K=1024 in 32 bf16 WMMA chunks. 256 waves -> 32 blocks.
// ---------------------------------------------------------------------------
__global__ void __launch_bounds__(256) tagproj_kernel(
    const float* __restrict__ hf, const float* __restrict__ hb,
    const float* __restrict__ w_tag, const float* __restrict__ b_tag,
    float* __restrict__ feats)
{
    int wave = blockIdx.x * 8 + (threadIdx.x >> 5);
    int lane = threadIdx.x & 31;
    int mt = wave;
    int m0 = mt * 16;
    int Ml = lane & 15, khalf = (lane >> 4) & 1;

    int  nB[2]     = { Ml, 16 + Ml };
    bool nvalid[2] = { true, (16 + Ml) < TAGS };
    const float* wrow[2];
    #pragma unroll
    for (int s = 0; s < 2; ++s)
        wrow[s] = w_tag + (long)(nvalid[s] ? nB[s] : 0) * HID;
    const float* arowf = hf + (long)(m0 + Ml) * H;
    const float* arowb = hb + (long)(m0 + Ml) * H;

    v8f acc[2] = {};
    for (int kt = 0; kt < 32; ++kt) {
        const float* arow = (kt < 16) ? arowf : arowb;     // lstm_out = [hf | hb]
        int kloc = (kt & 15) * 32;
        const float* a0 = arow + kloc + khalf * 16;
        const float* a1 = a0 + 8;
        v16bf A = pack_tile(*(const float4*)a0, *(const float4*)(a0 + 4),
                            *(const float4*)a1, *(const float4*)(a1 + 4));
        int kabs = kt * 32 + khalf * 16;
        #pragma unroll
        for (int s = 0; s < 2; ++s) {
            v16bf B;
            if (nvalid[s]) {
                const float* bp = wrow[s] + kabs;
                B = pack_tile(*(const float4*)bp,       *(const float4*)(bp + 4),
                              *(const float4*)(bp + 8), *(const float4*)(bp + 12));
            } else {
                v8u z = {};
                B = __builtin_bit_cast(v16bf, z);
            }
            acc[s] = __builtin_amdgcn_wmma_f32_16x16x32_bf16(
                         false, A, false, B, (short)0, acc[s], false, false);
        }
    }
    #pragma unroll
    for (int s = 0; s < 2; ++s) {
        if (!nvalid[s]) continue;
        float bias = b_tag[nB[s]];
        #pragma unroll
        for (int v = 0; v < 8; ++v) {
            int m = m0 + v + khalf * 8;
            feats[(long)m * TAGS + nB[s]] = acc[s][v] + bias;
        }
    }
}

// ---------------------------------------------------------------------------
// Kernel D: Viterbi forward pass (max-plus, 24 tags, one wave32) + backtrace.
// ---------------------------------------------------------------------------
__global__ void __launch_bounds__(32) viterbi_kernel(
    const float* __restrict__ feats, const float* __restrict__ trans,
    int* __restrict__ bp, float* __restrict__ score_out, int* __restrict__ path_out)
{
    __shared__ float trans_s[TAGS * TAGS];
    __shared__ float fvA[32], fvB[32], term[TAGS];
    int l = threadIdx.x;
    for (int i = l; i < TAGS * TAGS; i += 32) trans_s[i] = trans[i];
    fvA[l] = (l == START_TAG) ? 0.0f : NEGV;
    __syncthreads();

    float* fva = fvA; float* fvb = fvB;
    for (int t = 0; t < SEQ; ++t) {
        float best = -3.4e38f; int barg = 0;
        if (l < TAGS) {
            #pragma unroll 4
            for (int p = 0; p < TAGS; ++p) {
                float sc = trans_s[l * TAGS + p] + fva[p];
                if (sc > best) { best = sc; barg = p; }   // first-max == jnp.argmax
            }
            best += feats[t * TAGS + l];
        }
        __syncthreads();
        if (l < TAGS) { fvb[l] = best; bp[t * TAGS + l] = barg; }
        __syncthreads();
        float* tmp = fva; fva = fvb; fvb = tmp;
    }

    if (l < TAGS) term[l] = fva[l] + trans_s[STOP_TAG * TAGS + l];
    __syncthreads();
    if (l == 0) {
        int bi = 0; float bs = term[0];
        for (int n = 1; n < TAGS; ++n) if (term[n] > bs) { bs = term[n]; bi = n; }
        score_out[0] = bs;
        int tag = bi;
        path_out[SEQ - 1] = bi;
        for (int t = SEQ - 1; t >= 1; --t) {
            tag = bp[t * TAGS + tag];
            path_out[t - 1] = tag;
        }
    }
}

// ---------------------------------------------------------------------------
extern "C" void kernel_launch(void* const* d_in, const int* in_sizes, int n_in,
                              void* d_out, int out_size, void* d_ws, size_t ws_size,
                              hipStream_t stream) {
    (void)in_sizes; (void)n_in; (void)out_size; (void)ws_size;
    const int*   sent   = (const int*)  d_in[0];
    const float* emb    = (const float*)d_in[1];
    const float* w_ih_f = (const float*)d_in[2];
    const float* w_hh_f = (const float*)d_in[3];
    const float* b_ih_f = (const float*)d_in[4];
    const float* b_hh_f = (const float*)d_in[5];
    const float* w_ih_b = (const float*)d_in[6];
    const float* w_hh_b = (const float*)d_in[7];
    const float* b_ih_b = (const float*)d_in[8];
    const float* b_hh_b = (const float*)d_in[9];
    const float* w_tag  = (const float*)d_in[10];
    const float* b_tag  = (const float*)d_in[11];
    const float* trans  = (const float*)d_in[12];
    const float* h0     = (const float*)d_in[13];
    const float* c0     = (const float*)d_in[14];

    // workspace layout (floats)
    float* ws     = (float*)d_ws;
    float* xg_f   = ws;                       // 4096*2048
    float* xg_b   = xg_f + (size_t)SEQ * G4;
    float* hf     = xg_b + (size_t)SEQ * G4;  // 4096*512
    float* hb     = hf   + (size_t)SEQ * H;
    float* feats  = hb   + (size_t)SEQ * H;   // 4096*24
    int*   bp     = (int*)(feats + (size_t)SEQ * TAGS);
    float* h_glob = (float*)(bp + (size_t)SEQ * TAGS);  // 2*512
    int*   bar    = (int*)(h_glob + 2 * H);             // 2 counters

    (void)hipMemsetAsync(bar, 0, 2 * sizeof(int), stream);  // capture-safe reset

    // A: embedding + input projection (WMMA bf16, b128 loads, 4 N-tiles/wave)
    embed_proj_kernel<<<2048, 256, 0, stream>>>(
        sent, emb, w_ih_f, b_ih_f, b_hh_f, w_ih_b, b_ih_b, b_hh_b, xg_f, xg_b);

    // B: serial scan, 16 cooperating WGs per direction, W_hh in LDS via TDM
    size_t lds_bytes = (size_t)(128 * 520 + H + 128) * sizeof(float);
    lstm_scan_kernel<<<32, 256, lds_bytes, stream>>>(
        xg_f, xg_b, w_hh_f, w_hh_b, h0, c0, hf, hb, h_glob, bar);

    // C: tag projection (WMMA bf16)
    tagproj_kernel<<<32, 256, 0, stream>>>(hf, hb, w_tag, b_tag, feats);

    // D: Viterbi + backtrace; d_out = [score_f32, path_i32[4096]]
    viterbi_kernel<<<1, 32, 0, stream>>>(feats, trans, bp,
                                         (float*)d_out, ((int*)d_out) + 1);
}
